// ConformerAttention_42795054137612
// MI455X (gfx1250) — matrix operand order repaired and checked by reading, not compiled, not run on hardware
//
#include <hip/hip_runtime.h>
#include <hip/hip_bf16.h>

#define D_MODEL 512
#define S_LEN   1024
#define B_SZ    8
#define H_NUM   8
#define HD      64
#define PLEN    2047   // 2*S-1

typedef __attribute__((ext_vector_type(16))) __bf16 v16bf;
typedef __attribute__((ext_vector_type(8)))  __bf16 v8bf;
typedef __attribute__((ext_vector_type(8)))  float  v8f;

union FragU { v16bf v; v8bf h[2]; };

__device__ __forceinline__ unsigned short f2bfu(float f) {
    unsigned int u = __builtin_bit_cast(unsigned int, f);
    return (unsigned short)((u + 0x7FFFu + ((u >> 16) & 1u)) >> 16);
}

__device__ __forceinline__ v8f vzero8() {
    v8f z;
#pragma unroll
    for (int i = 0; i < 8; ++i) z[i] = 0.0f;
    return z;
}

// A-fragment (16x32 bf16): elements 0..7 = K klo..klo+7, 8..15 = K klo+16..klo+23
__device__ __forceinline__ v16bf load_frag_a(const unsigned short* base, int klo) {
    FragU f;
    f.h[0] = *(const v8bf*)(base + klo);
    f.h[1] = *(const v8bf*)(base + klo + 16);
    return f.v;
}
// B-fragment (32x16 bf16): elements 0..7 = K kloB..kloB+7, 8..15 = K kloB+16..kloB+23
__device__ __forceinline__ v16bf load_frag_b(const unsigned short* base, int kloB) {
    FragU f;
    f.h[0] = *(const v8bf*)(base + kloB);
    f.h[1] = *(const v8bf*)(base + kloB + 16);
    return f.v;
}

__device__ __forceinline__ v8f wmma_bf16(v16bf a, v16bf b, v8f c) {
    return __builtin_amdgcn_wmma_f32_16x16x32_bf16(false, a, false, b, (short)0, c, false, false);
}

// ---------------------------------------------------------------- cvt f32->bf16
__global__ void cvt_kernel(const float* __restrict__ in, unsigned short* __restrict__ out, int n) {
    int i = blockIdx.x * blockDim.x + threadIdx.x;
    if (i < n) out[i] = f2bfu(in[i]);
}

// ---------------------------------------------------------------- LayerNorm -> bf16
__global__ __launch_bounds__(128) void ln_kernel(const float* __restrict__ x,
                                                 const float* __restrict__ w,
                                                 const float* __restrict__ b,
                                                 unsigned short* __restrict__ out) {
    __shared__ float sm[4];
    int row = blockIdx.x;
    const float* xr = x + (size_t)row * D_MODEL;
    float v[4];
    float s = 0.f;
#pragma unroll
    for (int i = 0; i < 4; ++i) { v[i] = xr[threadIdx.x + i * 128]; s += v[i]; }
#pragma unroll
    for (int o = 16; o > 0; o >>= 1) s += __shfl_xor(s, o, 32);
    int lane = threadIdx.x & 31, wv = threadIdx.x >> 5;
    if (lane == 0) sm[wv] = s;
    __syncthreads();
    float mu = (sm[0] + sm[1] + sm[2] + sm[3]) * (1.0f / D_MODEL);
    float s2 = 0.f;
#pragma unroll
    for (int i = 0; i < 4; ++i) { float d = v[i] - mu; s2 += d * d; }
#pragma unroll
    for (int o = 16; o > 0; o >>= 1) s2 += __shfl_xor(s2, o, 32);
    __syncthreads();
    if (lane == 0) sm[wv] = s2;
    __syncthreads();
    float rstd = rsqrtf((sm[0] + sm[1] + sm[2] + sm[3]) * (1.0f / D_MODEL) + 1e-5f);
#pragma unroll
    for (int i = 0; i < 4; ++i) {
        int c = threadIdx.x + i * 128;
        out[(size_t)row * D_MODEL + c] = f2bfu((v[i] - mu) * rstd * w[c] + b[c]);
    }
}

// ---------------------------------------------------------------- generic WMMA GEMM
// C[m,n] = sum_k A[m,k]*W[n,k]  (+bias). Block: 8 waves, each wave 32(M)x64(N).
// Software pipelined with explicit ping-pong buffers (2x unrolled k loop, K % 64 == 0)
// so the loop-carried fragments are load destinations (no v_mov rotation).
// mode 0: bf16 out at [B,H,S,hd] (out1 += extra1[n], optional out2 += extra2[n])
// mode 1: bf16 out transposed [B,H,hd,S]
// mode 2: bf16 out at [H,PLEN,hd]
// mode 3: f32 out = v + residual  (final projection)
__global__ __launch_bounds__(256) void gemm_kernel(
    const unsigned short* __restrict__ A, const unsigned short* __restrict__ W,
    int M, int K,
    const float* __restrict__ bias, const float* __restrict__ extra1, const float* __restrict__ extra2,
    unsigned short* __restrict__ out1, unsigned short* __restrict__ out2,
    const float* __restrict__ residual, float* __restrict__ out32, int mode) {
    int lane = threadIdx.x & 31, wave = threadIdx.x >> 5;
    int lid = lane & 15, hi = (lane >> 4) & 1;
    int M0 = blockIdx.x * 256 + wave * 32;
    int N0 = blockIdx.y * 64;
    int kloA = hi ? 8 : 0, kloB = hi ? 8 : 0;
    int r0 = M0 + lid;      if (r0 > M - 1) r0 = M - 1;
    int r1 = M0 + 16 + lid; if (r1 > M - 1) r1 = M - 1;
    const unsigned short* Ar0 = A + (size_t)r0 * K;
    const unsigned short* Ar1 = A + (size_t)r1 * K;
    const unsigned short* Wr[4];
#pragma unroll
    for (int t = 0; t < 4; ++t) Wr[t] = W + (size_t)(N0 + t * 16 + lid) * K;

    v8f acc[2][4];
#pragma unroll
    for (int mt = 0; mt < 2; ++mt)
#pragma unroll
        for (int t = 0; t < 4; ++t) acc[mt][t] = vzero8();

    // prologue: fill buffer A (k = 0)
    v16bf a0A = load_frag_a(Ar0, kloA);
    v16bf a1A = load_frag_a(Ar1, kloA);
    v16bf bA[4];
#pragma unroll
    for (int t = 0; t < 4; ++t) bA[t] = load_frag_b(Wr[t], kloB);

    for (int kk = 0; kk < K; kk += 64) {
        // fill buffer B (k = kk+32) while buffer A feeds the WMMAs
        int k1 = kk + 32;
        v16bf a0B = load_frag_a(Ar0 + k1, kloA);
        v16bf a1B = load_frag_a(Ar1 + k1, kloA);
        v16bf bB[4];
#pragma unroll
        for (int t = 0; t < 4; ++t) bB[t] = load_frag_b(Wr[t] + k1, kloB);
#pragma unroll
        for (int t = 0; t < 4; ++t) {
            acc[0][t] = wmma_bf16(a0A, bA[t], acc[0][t]);
            acc[1][t] = wmma_bf16(a1A, bA[t], acc[1][t]);
        }
        // refill buffer A (k = kk+64, wrapped on last step) while buffer B feeds WMMAs
        int k2 = (kk + 64 < K) ? kk + 64 : 0;
        a0A = load_frag_a(Ar0 + k2, kloA);
        a1A = load_frag_a(Ar1 + k2, kloA);
#pragma unroll
        for (int t = 0; t < 4; ++t) bA[t] = load_frag_b(Wr[t] + k2, kloB);
#pragma unroll
        for (int t = 0; t < 4; ++t) {
            acc[0][t] = wmma_bf16(a0B, bB[t], acc[0][t]);
            acc[1][t] = wmma_bf16(a1B, bB[t], acc[1][t]);
        }
    }

#pragma unroll
    for (int t = 0; t < 4; ++t) {
        int n = N0 + t * 16 + lid;
        float bs = bias ? bias[n] : 0.f;
        float e1 = extra1 ? extra1[n] : 0.f;
        float e2 = extra2 ? extra2[n] : 0.f;
#pragma unroll
        for (int mt = 0; mt < 2; ++mt) {
#pragma unroll
            for (int r = 0; r < 8; ++r) {
                int m = M0 + mt * 16 + r + hi * 8;
                if (m >= M) continue;
                float v = acc[mt][t][r] + bs;
                if (mode == 0) {
                    int bb = m >> 10, s = m & 1023;
                    int hh = n >> 6, d = n & 63;
                    size_t o = (((size_t)(bb * H_NUM + hh) * S_LEN) + s) * HD + d;
                    out1[o] = f2bfu(v + e1);
                    if (out2) out2[o] = f2bfu(v + e2);
                } else if (mode == 1) {
                    int bb = m >> 10, s = m & 1023;
                    int hh = n >> 6, d = n & 63;
                    size_t o = (((size_t)(bb * H_NUM + hh) * HD) + d) * S_LEN + s;
                    out1[o] = f2bfu(v);
                } else if (mode == 2) {
                    int hh = n >> 6, d = n & 63;
                    size_t o = ((size_t)hh * PLEN + m) * HD + d;
                    out1[o] = f2bfu(v);
                } else {
                    size_t o = (size_t)m * D_MODEL + n;
                    out32[o] = v + residual[o];
                }
            }
        }
    }
}

// ---------------------------------------------------------------- fused rel-pos flash attention
// grid: B*H*(S/128), block: 256 (8 waves). Wave owns 16 q rows, streams k in steps of 32.
__global__ __launch_bounds__(256) void attn_kernel(
    const unsigned short* __restrict__ qu, const unsigned short* __restrict__ qv,
    const unsigned short* __restrict__ kb, const unsigned short* __restrict__ vt,
    const unsigned short* __restrict__ pb, unsigned short* __restrict__ outb) {
    int lane = threadIdx.x & 31, wave = threadIdx.x >> 5;
    int lid = lane & 15, hi = (lane >> 4) & 1;
    int bid = blockIdx.x;
    int qt = bid & 7, h = (bid >> 3) & 7, b = bid >> 6;
    int q0 = qt * 128 + wave * 16;
    int bh = b * H_NUM + h;
    const unsigned short* quB = qu + (size_t)bh * S_LEN * HD;
    const unsigned short* qvB = qv + (size_t)bh * S_LEN * HD;
    const unsigned short* kB  = kb + (size_t)bh * S_LEN * HD;
    const unsigned short* vB  = vt + (size_t)bh * HD * S_LEN;
    const unsigned short* pB  = pb + (size_t)h * PLEN * HD;

    __shared__ float Tls[8][16 * 48];            // pos-score band, per wave
    __shared__ unsigned short Pls[8][16 * 32];   // probs (bf16) for C/D->A transpose
    float* T = Tls[wave];
    unsigned short* PL = Pls[wave];

    int kloA = hi ? 8 : 0, kloB = hi ? 8 : 0;

    // Q fragments are invariant over the k loop: preload (d chunks 0..31, 32..63)
    const unsigned short* qrow = quB + (size_t)(q0 + lid) * HD;
    const unsigned short* qrow2 = qvB + (size_t)(q0 + lid) * HD;
    v16bf au0 = load_frag_a(qrow, kloA);
    v16bf au1 = load_frag_a(qrow + 32, kloA);
    v16bf av0 = load_frag_a(qrow2, kloA);
    v16bf av1 = load_frag_a(qrow2 + 32, kloA);

    v8f O[4];
#pragma unroll
    for (int t = 0; t < 4; ++t) O[t] = vzero8();
    float m_r[8], l_r[8];
#pragma unroll
    for (int r = 0; r < 8; ++r) { m_r[r] = -3.0e38f; l_r[r] = 0.f; }

    const float scale = 0.125f; // 1/sqrt(64)

    for (int k0 = 0; k0 < S_LEN; k0 += 32) {
        // ---- batch all score-side fragment loads (one clause, waits amortize)
        v16bf kf[2][2];
#pragma unroll
        for (int t = 0; t < 2; ++t) {
            const unsigned short* kp = kB + (size_t)(k0 + t * 16 + lid) * HD;
            kf[t][0] = load_frag_b(kp, kloB);
            kf[t][1] = load_frag_b(kp + 32, kloB);
        }
        int base = k0 - q0 + (S_LEN - 1) - 15;
        v16bf pf3[3][2];
#pragma unroll
        for (int t = 0; t < 3; ++t) {
            int pc = base + t * 16 + lid;
            pc = pc < 0 ? 0 : (pc > PLEN - 1 ? PLEN - 1 : pc);
            const unsigned short* pp = pB + (size_t)pc * HD;
            pf3[t][0] = load_frag_b(pp, kloB);
            pf3[t][1] = load_frag_b(pp + 32, kloB);
        }
        // ---- content scores: two 16x16 tiles (k cols k0..k0+31)
        v8f sc[2];
#pragma unroll
        for (int t = 0; t < 2; ++t) {
            v8f a = vzero8();
            a = wmma_bf16(au0, kf[t][0], a);
            a = wmma_bf16(au1, kf[t][1], a);
            sc[t] = a;
        }
        // ---- positional band: 3 tiles covering p = base .. base+47
#pragma unroll
        for (int t = 0; t < 3; ++t) {
            v8f a = vzero8();
            a = wmma_bf16(av0, pf3[t][0], a);
            a = wmma_bf16(av1, pf3[t][1], a);
#pragma unroll
            for (int r = 0; r < 8; ++r)
                T[(r + hi * 8) * 48 + t * 16 + lid] = a[r];
        }
        // ---- issue V fragment loads now; they complete under the softmax VALU work
        v16bf vf[4];
#pragma unroll
        for (int t = 0; t < 4; ++t)
            vf[t] = load_frag_b(vB + (size_t)(t * 16 + lid) * S_LEN + k0, kloB);

        // ---- gather rel-shifted pos, combine, streaming softmax
        float s0[8], s1[8], nm[8];
#pragma unroll
        for (int r = 0; r < 8; ++r) {
            int qq = r + hi * 8;
            float p0 = T[qq * 48 + (lid - qq + 15)];
            float p1 = T[qq * 48 + (16 + lid - qq + 15)];
            s0[r] = (sc[0][r] + p0) * scale;
            s1[r] = (sc[1][r] + p1) * scale;
            float mx = fmaxf(s0[r], s1[r]);
#pragma unroll
            for (int o = 1; o < 16; o <<= 1) mx = fmaxf(mx, __shfl_xor(mx, o, 32));
            nm[r] = fmaxf(m_r[r], mx);
        }
        float alpha[8];
#pragma unroll
        for (int r = 0; r < 8; ++r) {
            alpha[r] = __expf(m_r[r] - nm[r]);
            m_r[r] = nm[r];
            float e0 = __expf(s0[r] - nm[r]);
            float e1 = __expf(s1[r] - nm[r]);
            int qq = r + hi * 8;
            PL[qq * 32 + lid] = f2bfu(e0);
            PL[qq * 32 + 16 + lid] = f2bfu(e1);
            float rs = e0 + e1;
#pragma unroll
            for (int o = 1; o < 16; o <<= 1) rs += __shfl_xor(rs, o, 32);
            l_r[r] = l_r[r] * alpha[r] + rs;
        }
#pragma unroll
        for (int t = 0; t < 4; ++t)
#pragma unroll
            for (int r = 0; r < 8; ++r) O[t][r] *= alpha[r];

        // ---- probs as A-fragment (transpose via LDS), accumulate O += P @ V
        FragU pfr;
        pfr.h[0] = *(const v8bf*)&PL[lid * 32 + kloA];
        pfr.h[1] = *(const v8bf*)&PL[lid * 32 + kloA + 16];
#pragma unroll
        for (int t = 0; t < 4; ++t)
            O[t] = wmma_bf16(pfr.v, vf[t], O[t]);
    }

    // ---- finalize: normalize and store bf16 at [B,S,H,hd]
    unsigned short* ob = outb + (((size_t)b * S_LEN + q0) * H_NUM + h) * HD;
#pragma unroll
    for (int r = 0; r < 8; ++r) {
        float inv = 1.0f / l_r[r];
        int srow = r + hi * 8;
#pragma unroll
        for (int t = 0; t < 4; ++t)
            ob[(size_t)srow * H_NUM * HD + t * 16 + lid] = f2bfu(O[t][r] * inv);
    }
}

// ----------------------------------------------------------------
extern "C" void kernel_launch(void* const* d_in, const int* in_sizes, int n_in,
                              void* d_out, int out_size, void* d_ws, size_t ws_size,
                              hipStream_t stream) {
    (void)in_sizes; (void)n_in; (void)out_size; (void)ws_size;
    const float* x      = (const float*)d_in[0];
    const float* posemb = (const float*)d_in[1];
    const float* ln_w   = (const float*)d_in[2];
    const float* ln_b   = (const float*)d_in[3];
    const float* Wq = (const float*)d_in[4];  const float* bq = (const float*)d_in[5];
    const float* Wk = (const float*)d_in[6];  const float* bk = (const float*)d_in[7];
    const float* Wv = (const float*)d_in[8];  const float* bv = (const float*)d_in[9];
    const float* Wo = (const float*)d_in[10]; const float* bo = (const float*)d_in[11];
    const float* Wp = (const float*)d_in[12];
    const float* pbu = (const float*)d_in[13];
    const float* pbv = (const float*)d_in[14];
    float* out = (float*)d_out;

    const size_t WEl = (size_t)D_MODEL * D_MODEL;       // 262144
    const size_t PosEl = (size_t)PLEN * D_MODEL;        // 1048064
    const size_t HEl = (size_t)B_SZ * S_LEN * D_MODEL;  // 4194304
    const size_t PprEl = (size_t)H_NUM * PLEN * HD;     // 1048064

    unsigned short* ws = (unsigned short*)d_ws;
    size_t off = 0;
    auto take = [&](size_t n) { unsigned short* p = ws + off; off += (n + 127) & ~(size_t)127; return p; };
    unsigned short* wqb = take(WEl);
    unsigned short* wkb = take(WEl);
    unsigned short* wvb = take(WEl);
    unsigned short* wob = take(WEl);
    unsigned short* wpb = take(WEl);
    unsigned short* posb = take(PosEl);
    unsigned short* hb  = take(HEl);   // reused as attention output later
    unsigned short* qub = take(HEl);
    unsigned short* qvb = take(HEl);
    unsigned short* kbf = take(HEl);
    unsigned short* vtb = take(HEl);
    unsigned short* pprj = take(PprEl);

    // weights / pos_emb -> bf16
    int cb = 256;
    cvt_kernel<<<(int)((WEl + cb - 1) / cb), cb, 0, stream>>>(Wq, wqb, (int)WEl);
    cvt_kernel<<<(int)((WEl + cb - 1) / cb), cb, 0, stream>>>(Wk, wkb, (int)WEl);
    cvt_kernel<<<(int)((WEl + cb - 1) / cb), cb, 0, stream>>>(Wv, wvb, (int)WEl);
    cvt_kernel<<<(int)((WEl + cb - 1) / cb), cb, 0, stream>>>(Wo, wob, (int)WEl);
    cvt_kernel<<<(int)((WEl + cb - 1) / cb), cb, 0, stream>>>(Wp, wpb, (int)WEl);
    cvt_kernel<<<(int)((PosEl + cb - 1) / cb), cb, 0, stream>>>(posemb, posb, (int)PosEl);

    // LayerNorm
    ln_kernel<<<B_SZ * S_LEN, 128, 0, stream>>>(x, ln_w, ln_b, hb);

    // Projections (block tile 256M x 64N)
    dim3 g1((B_SZ * S_LEN) / 256, D_MODEL / 64);
    gemm_kernel<<<g1, 256, 0, stream>>>(hb, wqb, B_SZ * S_LEN, D_MODEL, bq, pbu, pbv,
                                        qub, qvb, nullptr, nullptr, 0);
    gemm_kernel<<<g1, 256, 0, stream>>>(hb, wkb, B_SZ * S_LEN, D_MODEL, bk, nullptr, nullptr,
                                        kbf, nullptr, nullptr, nullptr, 0);
    gemm_kernel<<<g1, 256, 0, stream>>>(hb, wvb, B_SZ * S_LEN, D_MODEL, bv, nullptr, nullptr,
                                        vtb, nullptr, nullptr, nullptr, 1);
    dim3 g2((PLEN + 255) / 256, D_MODEL / 64);
    gemm_kernel<<<g2, 256, 0, stream>>>(posb, wpb, PLEN, D_MODEL, nullptr, nullptr, nullptr,
                                        pprj, nullptr, nullptr, nullptr, 2);

    // Fused rel-pos flash attention (writes into hb, which is free now)
    attn_kernel<<<B_SZ * H_NUM * (S_LEN / 128), 256, 0, stream>>>(qub, qvb, kbf, vtb, pprj, hb);

    // Output projection + residual -> fp32
    gemm_kernel<<<g1, 256, 0, stream>>>(hb, wob, B_SZ * S_LEN, D_MODEL, bo, nullptr, nullptr,
                                        nullptr, nullptr, x, out, 3);
}
